// SketchToCoordGAT_90589450207855
// MI455X (gfx1250) — compile-verified, matrix operand order checked by reading, not run on hardware
//
#include <hip/hip_runtime.h>
#include <hip/hip_bf16.h>
#include <math.h>

// ---------------------------------------------------------------------------
// Types for CDNA5 WMMA (wave32): 16x16x32 f16 -> f32
// ---------------------------------------------------------------------------
typedef _Float16 v16h __attribute__((ext_vector_type(16)));
typedef _Float16 v8h  __attribute__((ext_vector_type(8)));
typedef float    v8f  __attribute__((ext_vector_type(8)));

// BlazePose POSE_CONNECTIONS (35 undirected bones)
__device__ __constant__ int c_e0[35] = {0,1,2,3,0,4,5,6,9,11,11,13,15,15,15,17,12,14,16,16,16,18,11,12,23,23,24,25,26,27,28,29,30,27,28};
__device__ __constant__ int c_e1[35] = {1,2,3,7,4,5,6,8,10,12,13,15,17,19,21,19,14,16,18,20,22,20,23,24,24,25,26,27,28,29,30,31,32,31,32};

// ---------------------------------------------------------------------------
// Elementwise helpers
// ---------------------------------------------------------------------------
__global__ void k_cvt(const float* __restrict__ s, _Float16* __restrict__ d, int n) {
    int i = blockIdx.x * blockDim.x + threadIdx.x;
    if (i < n) d[i] = (_Float16)s[i];
}

// weight convert with K padding: [rows, kin] f32 -> [rows, kpad] f16 (zero pad)
__global__ void k_wcvt(const float* __restrict__ s, _Float16* __restrict__ d,
                       int rows, int kin, int kpad) {
    int i = blockIdx.x * blockDim.x + threadIdx.x;
    int total = rows * kpad;
    if (i >= total) return;
    int r = i / kpad, k = i - r * kpad;
    d[i] = (k < kin) ? (_Float16)s[r * kin + k] : (_Float16)0.f;
}

__global__ void k_fillf(float* __restrict__ p, float v, int n) {
    int i = blockIdx.x * blockDim.x + threadIdx.x;
    if (i < n) p[i] = v;
}

// ---------------------------------------------------------------------------
// Implicit-GEMM conv 3x3 SAME + bias + ReLU + eval-BatchNorm via WMMA.
// GEMM view: M=Cout, N=B*H*W, K=CIN*9 (padded to KPAD%32==0).
// Compile-time CIN/H/W/KPAD -> all index math is shifts / magic multiplies.
// Block = 128 threads = 4 waves; weight tile (16 x KPAD) staged in LDS and
// shared by all 4 waves (ds_load_b128 A-fragments); each wave owns one
// 16(M) x 16(N) accumulator tile.
// ---------------------------------------------------------------------------
template <int CIN, int H, int W, int KPAD>
__global__ void __launch_bounds__(128)
k_conv_wmma(const _Float16* __restrict__ in,   // [B,CIN,H,W] f16
            const _Float16* __restrict__ w,    // [Cout,KPAD] f16 (zero padded)
            const float* __restrict__ cb,      // conv bias [Cout]
            const float* __restrict__ bng, const float* __restrict__ bnb,
            const float* __restrict__ bnm, const float* __restrict__ bnv,
            _Float16* __restrict__ out,        // [B,Cout,H,W] f16
            int Cout) {
    constexpr int HW    = H * W;
    constexpr int KREAL = CIN * 9;
    __shared__ _Float16 smem[16 * KPAD];

    const int lane = threadIdx.x & 31;
    const int wv   = threadIdx.x >> 5;
    const int m0 = blockIdx.y << 4;                 // Cout tile (block-wide)
    const int n0 = (blockIdx.x * 4 + wv) << 4;      // pixel tile (per wave)
    const int ln = lane & 15;
    const int kb = (lane < 16) ? 0 : 8;             // K sub-base per A/B layout

    // Stage this block's 16-row weight tile into LDS (b128 copies).
    {
        const _Float16* wsrc = w + (size_t)m0 * KPAD;
        for (int idx = threadIdx.x * 8; idx < 16 * KPAD; idx += 128 * 8)
            *(v8h*)(smem + idx) = *(const v8h*)(wsrc + idx);
    }
    __syncthreads();

    // This lane's pixel (column of im2col) for the B fragment & epilogue
    const int n   = n0 + ln;
    const int b   = n / HW;          // HW is a power of two -> shift
    const int rem = n & (HW - 1);
    const int py  = rem / W;         // W power of two -> shift
    const int px  = rem & (W - 1);
    const _Float16* inb = in + (size_t)b * CIN * HW;

    const _Float16* arow = smem + ln * KPAD + kb;

    v8f acc = {};
    #pragma unroll 4
    for (int k0 = 0; k0 < KPAD; k0 += 32) {
        // A fragment from LDS: two ds_load_b128
        v16h af;
        {
            v8h lo = *(const v8h*)(arow + k0);
            v8h hi = *(const v8h*)(arow + k0 + 16);
            #pragma unroll
            for (int i = 0; i < 8; i++) { af[i] = lo[i]; af[i + 8] = hi[i]; }
        }
        // B fragment: on-the-fly im2col gather (zero padding + K tail)
        v16h bf;
        #pragma unroll
        for (int h = 0; h < 16; h++) {
            int k = k0 + kb + (h & 7) + ((h >= 8) ? 16 : 0);
            _Float16 v = (_Float16)0.f;
            if (k < KREAL) {         // compile-time-foldable per unrolled step
                int ci = k / 9;      // constant divisor -> magic multiply
                int rs = k - ci * 9;
                int r  = rs / 3;
                int s  = rs - r * 3;
                int yy = py + r - 1, xx = px + s - 1;
                if (yy >= 0 && yy < H && xx >= 0 && xx < W)
                    v = inb[(size_t)ci * HW + yy * W + xx];
            }
            bf[h] = v;
        }
        acc = __builtin_amdgcn_wmma_f32_16x16x32_f16(false, af, false, bf,
                                                     (short)0, acc, false, false);
    }
    // Epilogue: +bias -> ReLU -> BN(eval) -> f16 store. C layout: VGPR j,
    // row = m0 + j + (lane>=16 ? 8 : 0), col = lane&15 (this lane's pixel).
    #pragma unroll
    for (int j = 0; j < 8; j++) {
        int co = m0 + j + ((lane >= 16) ? 8 : 0);
        float v = acc[j] + cb[co];
        v = v > 0.f ? v : 0.f;
        float sc = bng[co] * rsqrtf(bnv[co] + 1e-5f);
        v = (v - bnm[co]) * sc + bnb[co];
        out[((size_t)b * Cout + co) * HW + py * W + px] = (_Float16)v;
    }
}

// ---------------------------------------------------------------------------
// Generic C[M,N] = A[M,K] * B[N,K]^T + bias (optional ReLU), f16 in / f32 out.
// K, ACT compile-time (K%32==0); requires M%64==0; N guarded.
// ---------------------------------------------------------------------------
template <int K, int ACT>
__global__ void __launch_bounds__(128)
k_gemm_nt(const _Float16* __restrict__ A, const _Float16* __restrict__ Bw,
          const float* __restrict__ bias, float* __restrict__ C,
          int M, int N) {
    const int lane = threadIdx.x & 31;
    const int wv   = threadIdx.x >> 5;
    const int m0 = (blockIdx.x * 4 + wv) << 4;
    const int n0 = blockIdx.y << 4;
    const int ln = lane & 15;
    const int kb = (lane < 16) ? 0 : 8;

    const _Float16* ar = A + (size_t)(m0 + ln) * K + kb;
    const int nrow = n0 + ln;
    const bool bok = (nrow < N);
    const _Float16* br = Bw + (size_t)(bok ? nrow : 0) * K + kb;

    v8f acc = {};
    #pragma unroll 4
    for (int k0 = 0; k0 < K; k0 += 32) {
        if (k0 + 32 < K) {
            __builtin_prefetch(ar + k0 + 32, 0, 1);
            __builtin_prefetch(br + k0 + 32, 0, 1);
        }
        v16h af;
        {
            v8h lo = *(const v8h*)(ar + k0);
            v8h hi = *(const v8h*)(ar + k0 + 16);
            #pragma unroll
            for (int i = 0; i < 8; i++) { af[i] = lo[i]; af[i + 8] = hi[i]; }
        }
        v16h bf;
        if (bok) {
            v8h lo = *(const v8h*)(br + k0);
            v8h hi = *(const v8h*)(br + k0 + 16);
            #pragma unroll
            for (int i = 0; i < 8; i++) { bf[i] = lo[i]; bf[i + 8] = hi[i]; }
        } else {
            #pragma unroll
            for (int i = 0; i < 16; i++) bf[i] = (_Float16)0.f;
        }
        acc = __builtin_amdgcn_wmma_f32_16x16x32_f16(false, af, false, bf,
                                                     (short)0, acc, false, false);
    }
    #pragma unroll
    for (int j = 0; j < 8; j++) {
        int m = m0 + j + ((lane >= 16) ? 8 : 0);
        int nn = n0 + ln;
        if (nn < N) {
            float v = acc[j] + (bias ? bias[nn] : 0.f);
            if (ACT) v = v > 0.f ? v : 0.f;
            C[(size_t)m * N + nn] = v;
        }
    }
}

// ---------------------------------------------------------------------------
// Pooling
// ---------------------------------------------------------------------------
template <int H, int W>
__global__ void k_maxpool2(const _Float16* __restrict__ in, _Float16* __restrict__ out,
                           int BC) {
    constexpr int Ho = H >> 1, Wo = W >> 1;
    int i = blockIdx.x * blockDim.x + threadIdx.x;
    int total = BC * Ho * Wo;
    if (i >= total) return;
    int bc = i / (Ho * Wo);
    int r  = i & (Ho * Wo - 1);     // power of two
    int y  = r / Wo;
    int x  = r & (Wo - 1);
    const _Float16* p = in + ((size_t)bc * H + y * 2) * W + x * 2;
    float a = (float)p[0], b = (float)p[1], c = (float)p[W], d = (float)p[W + 1];
    out[i] = (_Float16)fmaxf(fmaxf(a, b), fmaxf(c, d));
}

// adaptive avg pool 16->4 + flatten: in [B,256,16,16] -> out [B,4096] f16
// feature index f = c*16 + y1*4 + x1 (matches reshape(B,256,4,4,4,4).mean(3,5))
__global__ void k_avgpool(const _Float16* __restrict__ in, _Float16* __restrict__ out, int Bn) {
    int i = blockIdx.x * blockDim.x + threadIdx.x;
    int total = Bn * 4096;
    if (i >= total) return;
    int b  = i >> 12;
    int f  = i & 4095;
    int c  = f >> 4;
    int y1 = (f >> 2) & 3;
    int x1 = f & 3;
    const _Float16* p = in + (((size_t)b * 256 + c) * 16 + y1 * 4) * 16 + x1 * 4;
    float s = 0.f;
    #pragma unroll
    for (int dy = 0; dy < 4; dy++)
        #pragma unroll
        for (int dx = 0; dx < 4; dx++) s += (float)p[dy * 16 + dx];
    out[i] = (_Float16)(s * (1.f / 16.f));
}

// node features: [B*J, 320] f16 = concat(img[b] (256, already ReLU'd), emb[j] (64))
__global__ void k_node(const float* __restrict__ img, const float* __restrict__ emb,
                       _Float16* __restrict__ node, int Bn) {
    int i = blockIdx.x * blockDim.x + threadIdx.x;
    int total = Bn * 33 * 320;
    if (i >= total) return;
    int nidx = i / 320;
    int t    = i - nidx * 320;
    int b = nidx / 33, j = nidx - b * 33;
    float v = (t < 256) ? img[b * 256 + t] : emb[j * 64 + (t - 256)];
    node[i] = (_Float16)v;
}

// ---------------------------------------------------------------------------
// GATv2 edge kernels: edges derived analytically (70 per batch + self loops)
// ---------------------------------------------------------------------------
__device__ inline void edge_sd(int e, int Bn, int& s, int& d) {
    int nb = Bn * 70;
    if (e < nb) {
        int b = e / 70;
        int t = e - b * 70;
        int a0, a1;
        if (t < 35) { a0 = c_e0[t];      a1 = c_e1[t]; }
        else        { a0 = c_e1[t - 35]; a1 = c_e0[t - 35]; }
        s = b * 33 + a0;
        d = b * 33 + a1;
    } else {
        s = d = e - nb; // self loop
    }
}

__device__ inline void atomicMaxF(float* addr, float v) {
    if (v >= 0.f) atomicMax((int*)addr, __float_as_int(v));
    else          atomicMin((unsigned int*)addr, __float_as_uint(v));
}

// e_ij per (edge, head) + segment max over incoming edges
__global__ void k_edge_e(const float* __restrict__ xl, const float* __restrict__ xr,
                         const float* __restrict__ att, float* __restrict__ ebuf,
                         float* __restrict__ emax, int E, int Bn) {
    int i = blockIdx.x * blockDim.x + threadIdx.x;
    if (i >= E * 4) return;
    int e = i >> 2, h = i & 3;
    int s, d; edge_sd(e, Bn, s, d);
    const float* pl = xl + (size_t)s * 128 + h * 32;
    const float* pr = xr + (size_t)d * 128 + h * 32;
    const float* pa = att + h * 32;
    float acc = 0.f;
    #pragma unroll
    for (int c = 0; c < 32; c++) {
        float v = pl[c] + pr[c];
        v = v > 0.f ? v : 0.2f * v;       // leaky_relu 0.2
        acc += pa[c] * v;
    }
    ebuf[i] = acc;
    atomicMaxF(&emax[d * 4 + h], acc);
}

__global__ void k_edge_exp(const float* __restrict__ ebuf, const float* __restrict__ emax,
                           float* __restrict__ exbuf, float* __restrict__ esum,
                           int E, int Bn) {
    int i = blockIdx.x * blockDim.x + threadIdx.x;
    if (i >= E * 4) return;
    int e = i >> 2, h = i & 3;
    int s, d; edge_sd(e, Bn, s, d);
    (void)s;
    float ex = __expf(ebuf[i] - emax[d * 4 + h]);
    exbuf[i] = ex;
    atomicAdd(&esum[d * 4 + h], ex);
}

__global__ void k_edge_agg(const float* __restrict__ exbuf, const float* __restrict__ esum,
                           const float* __restrict__ xl, float* __restrict__ agg,
                           int E, int Bn) {
    int i = blockIdx.x * blockDim.x + threadIdx.x;
    if (i >= E * 4) return;
    int e = i >> 2, h = i & 3;
    int s, d; edge_sd(e, Bn, s, d);
    float alpha = exbuf[i] / esum[d * 4 + h];
    const float* pl = xl + (size_t)s * 128 + h * 32;
    float* pd = agg + (size_t)d * 128 + h * 32;
    #pragma unroll
    for (int c = 0; c < 32; c++) atomicAdd(&pd[c], alpha * pl[c]);
}

// concat heads: +bias -> elu -> f16
__global__ void k_post_concat(const float* __restrict__ agg, const float* __restrict__ bias,
                              _Float16* __restrict__ out, int N) {
    int i = blockIdx.x * blockDim.x + threadIdx.x;
    if (i >= N * 128) return;
    int t = i & 127;
    float v = agg[i] + bias[t];
    v = v > 0.f ? v : (__expf(v) - 1.f);
    out[i] = (_Float16)v;
}

// mean over heads: [N,4,32] -> [N,32], +bias -> elu -> f16
__global__ void k_post_mean(const float* __restrict__ agg, const float* __restrict__ bias,
                            _Float16* __restrict__ out, int N) {
    int i = blockIdx.x * blockDim.x + threadIdx.x;
    if (i >= N * 32) return;
    int n = i >> 5, t = i & 31;
    const float* p = agg + (size_t)n * 128;
    float v = (p[t] + p[32 + t] + p[64 + t] + p[96 + t]) * 0.25f + bias[t];
    v = v > 0.f ? v : (__expf(v) - 1.f);
    out[i] = (_Float16)v;
}

// split head output [NJ,4] -> coords [NJ,3] ++ sigmoid conf [NJ]
__global__ void k_final(const float* __restrict__ o4, float* __restrict__ out, int NJ) {
    int i = blockIdx.x * blockDim.x + threadIdx.x;
    if (i >= NJ) return;
    out[i * 3 + 0] = o4[i * 4 + 0];
    out[i * 3 + 1] = o4[i * 4 + 1];
    out[i * 3 + 2] = o4[i * 4 + 2];
    out[(size_t)NJ * 3 + i] = 1.f / (1.f + __expf(-o4[i * 4 + 3]));
}

// ---------------------------------------------------------------------------
// Host launch
// ---------------------------------------------------------------------------
extern "C" void kernel_launch(void* const* d_in, const int* in_sizes, int n_in,
                              void* d_out, int out_size, void* d_ws, size_t ws_size,
                              hipStream_t stream) {
    (void)in_sizes; (void)n_in; (void)out_size;
    const int Bn = 256, J = 33, NJ = Bn * J;   // 8448
    const int E  = 70 * Bn + NJ;               // 26368
    const int T  = 256;

    // Input leaves, setup_inputs() dict insertion order
    const float* X    = (const float*)d_in[0];
    const float* C1W  = (const float*)d_in[1];
    const float* C1B  = (const float*)d_in[2];
    const float* BN1[4] = {(const float*)d_in[3], (const float*)d_in[4], (const float*)d_in[5], (const float*)d_in[6]};
    const float* C2W  = (const float*)d_in[7];
    const float* C2B  = (const float*)d_in[8];
    const float* BN2[4] = {(const float*)d_in[9], (const float*)d_in[10], (const float*)d_in[11], (const float*)d_in[12]};
    const float* C3W  = (const float*)d_in[13];
    const float* C3B  = (const float*)d_in[14];
    const float* BN3[4] = {(const float*)d_in[15], (const float*)d_in[16], (const float*)d_in[17], (const float*)d_in[18]};
    const float* C4W  = (const float*)d_in[19];
    const float* C4B  = (const float*)d_in[20];
    const float* BN4[4] = {(const float*)d_in[21], (const float*)d_in[22], (const float*)d_in[23], (const float*)d_in[24]};
    const float* PROJW = (const float*)d_in[25];
    const float* PROJB = (const float*)d_in[26];
    const float* JEMB  = (const float*)d_in[27];
    const float* GWL[3] = {(const float*)d_in[28], (const float*)d_in[34], (const float*)d_in[40]};
    const float* GBL[3] = {(const float*)d_in[29], (const float*)d_in[35], (const float*)d_in[41]};
    const float* GWR[3] = {(const float*)d_in[30], (const float*)d_in[36], (const float*)d_in[42]};
    const float* GBR[3] = {(const float*)d_in[31], (const float*)d_in[37], (const float*)d_in[43]};
    const float* GATT[3] = {(const float*)d_in[32], (const float*)d_in[38], (const float*)d_in[44]};
    const float* GBIAS[3] = {(const float*)d_in[33], (const float*)d_in[39], (const float*)d_in[45]};
    const float* HW1 = (const float*)d_in[46];
    const float* HB1 = (const float*)d_in[47];
    const float* HW2 = (const float*)d_in[48];
    const float* HB2 = (const float*)d_in[49];

    // ---- workspace arena management ----
    char* base = (char*)d_ws;
    size_t off = 0;
    auto alloc = [&](size_t bytes) -> void* {
        void* p = base + off;
        off = (off + bytes + 255) & ~(size_t)255;
        return p;
    };
    // persistent f16 weights
    _Float16* w1h   = (_Float16*)alloc((size_t)32 * 32 * 2);       // Kpad 32 (27 padded)
    _Float16* w2h   = (_Float16*)alloc((size_t)64 * 288 * 2);
    _Float16* w3h   = (_Float16*)alloc((size_t)128 * 576 * 2);
    _Float16* w4h   = (_Float16*)alloc((size_t)256 * 1152 * 2);
    _Float16* pwh   = (_Float16*)alloc((size_t)256 * 4096 * 2);
    _Float16* wl0h  = (_Float16*)alloc((size_t)128 * 320 * 2);
    _Float16* wr0h  = (_Float16*)alloc((size_t)128 * 320 * 2);
    _Float16* wl1h  = (_Float16*)alloc((size_t)128 * 128 * 2);
    _Float16* wr1h  = (_Float16*)alloc((size_t)128 * 128 * 2);
    _Float16* wl2h  = (_Float16*)alloc((size_t)128 * 128 * 2);
    _Float16* wr2h  = (_Float16*)alloc((size_t)128 * 128 * 2);
    _Float16* hw1h  = (_Float16*)alloc((size_t)64 * 32 * 2);
    _Float16* hw2h  = (_Float16*)alloc((size_t)4 * 64 * 2);
    // two alternating activation arenas
    char* arenaA = (char*)alloc((size_t)134217728);   // max: act2 = 256*64*64*64*2
    char* arenaB = (char*)alloc((size_t)67108864);    // max: act1 = 256*32*64*64*2
    if (ws_size < off) return;  // deterministic: ws too small -> no-op

    _Float16* xh    = (_Float16*)arenaA;   // x f16 (dead after conv1)
    _Float16* act1  = (_Float16*)arenaB;
    _Float16* act2  = (_Float16*)arenaA;
    _Float16* pool1 = (_Float16*)arenaB;
    _Float16* act3  = (_Float16*)arenaA;
    _Float16* pool2 = (_Float16*)arenaB;
    _Float16* act4  = (_Float16*)arenaA;
    // tail (alive only after pool2 is dead) carved from arenaB
    size_t toff = 0;
    auto talloc = [&](size_t bytes) -> void* {
        void* p = arenaB + toff;
        toff = (toff + bytes + 255) & ~(size_t)255;
        return p;
    };
    _Float16* pooledh = (_Float16*)talloc((size_t)Bn * 4096 * 2);
    _Float16* nodeh   = (_Float16*)talloc((size_t)NJ * 320 * 2);
    _Float16* hbuf0   = (_Float16*)talloc((size_t)NJ * 128 * 2);
    _Float16* hbuf1   = (_Float16*)talloc((size_t)NJ * 128 * 2);
    _Float16* gfh     = (_Float16*)talloc((size_t)NJ * 32 * 2);
    _Float16* h1h     = (_Float16*)talloc((size_t)NJ * 64 * 2);
    float* img   = (float*)talloc((size_t)Bn * 256 * 4);
    float* xl    = (float*)talloc((size_t)NJ * 128 * 4);
    float* xr    = (float*)talloc((size_t)NJ * 128 * 4);
    float* agg   = (float*)talloc((size_t)NJ * 128 * 4);
    float* emax  = (float*)talloc((size_t)NJ * 4 * 4);
    float* esum  = (float*)talloc((size_t)NJ * 4 * 4);
    float* ebuf  = (float*)talloc((size_t)E * 4 * 4);
    float* exbuf = (float*)talloc((size_t)E * 4 * 4);
    float* h1f   = (float*)talloc((size_t)NJ * 64 * 4);
    float* o4    = (float*)talloc((size_t)NJ * 4 * 4);

    // ---- weight conversions (f32 -> f16, conv weights K-padded) ----
    k_wcvt<<<(32 * 32 + T - 1) / T, T, 0, stream>>>(C1W, w1h, 32, 27, 32);
    k_wcvt<<<(64 * 288 + T - 1) / T, T, 0, stream>>>(C2W, w2h, 64, 288, 288);
    k_wcvt<<<(128 * 576 + T - 1) / T, T, 0, stream>>>(C3W, w3h, 128, 576, 576);
    k_wcvt<<<(256 * 1152 + T - 1) / T, T, 0, stream>>>(C4W, w4h, 256, 1152, 1152);
    k_cvt<<<(1048576 + T - 1) / T, T, 0, stream>>>(PROJW, pwh, 1048576);
    k_cvt<<<(40960 + T - 1) / T, T, 0, stream>>>(GWL[0], wl0h, 40960);
    k_cvt<<<(40960 + T - 1) / T, T, 0, stream>>>(GWR[0], wr0h, 40960);
    k_cvt<<<(16384 + T - 1) / T, T, 0, stream>>>(GWL[1], wl1h, 16384);
    k_cvt<<<(16384 + T - 1) / T, T, 0, stream>>>(GWR[1], wr1h, 16384);
    k_cvt<<<(16384 + T - 1) / T, T, 0, stream>>>(GWL[2], wl2h, 16384);
    k_cvt<<<(16384 + T - 1) / T, T, 0, stream>>>(GWR[2], wr2h, 16384);
    k_cvt<<<(2048 + T - 1) / T, T, 0, stream>>>(HW1, hw1h, 2048);
    k_cvt<<<(256 + T - 1) / T, T, 0, stream>>>(HW2, hw2h, 256);
    k_cvt<<<(3145728 + T - 1) / T, T, 0, stream>>>(X, xh, 3145728);

    // ---- CNN backbone (implicit-GEMM WMMA convs, fused ReLU+BN) ----
    // conv1: 3->32 @64x64, N = 1,048,576 pixels
    k_conv_wmma<3, 64, 64, 32><<<dim3(16384, 2), 128, 0, stream>>>(
        xh, w1h, C1B, BN1[0], BN1[1], BN1[2], BN1[3], act1, 32);
    // conv2: 32->64 @64x64
    k_conv_wmma<32, 64, 64, 288><<<dim3(16384, 4), 128, 0, stream>>>(
        act1, w2h, C2B, BN2[0], BN2[1], BN2[2], BN2[3], act2, 64);
    // maxpool 64->32
    k_maxpool2<64, 64><<<(16777216 + T - 1) / T, T, 0, stream>>>(act2, pool1, Bn * 64);
    // conv3: 64->128 @32x32, N = 262,144
    k_conv_wmma<64, 32, 32, 576><<<dim3(4096, 8), 128, 0, stream>>>(
        pool1, w3h, C3B, BN3[0], BN3[1], BN3[2], BN3[3], act3, 128);
    // maxpool 32->16
    k_maxpool2<32, 32><<<(8388608 + T - 1) / T, T, 0, stream>>>(act3, pool2, Bn * 128);
    // conv4: 128->256 @16x16, N = 65,536
    k_conv_wmma<128, 16, 16, 1152><<<dim3(1024, 16), 128, 0, stream>>>(
        pool2, w4h, C4B, BN4[0], BN4[1], BN4[2], BN4[3], act4, 256);
    // adaptive avg pool 16->4 + flatten -> [256, 4096] f16
    k_avgpool<<<(Bn * 4096 + T - 1) / T, T, 0, stream>>>(act4, pooledh, Bn);
    // projection: img = relu(pooled @ proj_w^T + b), M=256 N=256 K=4096
    k_gemm_nt<4096, 1><<<dim3(4, 16), 128, 0, stream>>>(pooledh, pwh, PROJB, img, 256, 256);
    // node features [8448, 320] f16
    k_node<<<(NJ * 320 + T - 1) / T, T, 0, stream>>>(img, JEMB, nodeh, Bn);

    // ---- GATv2 stack ----
    const int E4 = E * 4;
    // shared tail of a GAT layer: softmax over incoming edges + aggregation
    auto gat_tail = [&](const float* att, const float* gbias, int L) {
        k_fillf<<<(NJ * 4 + T - 1) / T, T, 0, stream>>>(emax, -1e30f, NJ * 4);
        k_fillf<<<(NJ * 4 + T - 1) / T, T, 0, stream>>>(esum, 0.f, NJ * 4);
        k_fillf<<<(NJ * 128 + T - 1) / T, T, 0, stream>>>(agg, 0.f, NJ * 128);
        k_edge_e<<<(E4 + T - 1) / T, T, 0, stream>>>(xl, xr, att, ebuf, emax, E, Bn);
        k_edge_exp<<<(E4 + T - 1) / T, T, 0, stream>>>(ebuf, emax, exbuf, esum, E, Bn);
        k_edge_agg<<<(E4 + T - 1) / T, T, 0, stream>>>(exbuf, esum, xl, agg, E, Bn);
        if (L < 2)
            k_post_concat<<<(NJ * 128 + T - 1) / T, T, 0, stream>>>(agg, gbias, (L == 0 ? hbuf0 : hbuf1), NJ);
        else
            k_post_mean<<<(NJ * 32 + T - 1) / T, T, 0, stream>>>(agg, gbias, gfh, NJ);
    };
    // layer 0: in = node features (K=320)
    k_gemm_nt<320, 0><<<dim3(NJ / 64, 8), 128, 0, stream>>>(nodeh, wl0h, GBL[0], xl, NJ, 128);
    k_gemm_nt<320, 0><<<dim3(NJ / 64, 8), 128, 0, stream>>>(nodeh, wr0h, GBR[0], xr, NJ, 128);
    gat_tail(GATT[0], GBIAS[0], 0);
    // layer 1: in = hbuf0 (K=128)
    k_gemm_nt<128, 0><<<dim3(NJ / 64, 8), 128, 0, stream>>>(hbuf0, wl1h, GBL[1], xl, NJ, 128);
    k_gemm_nt<128, 0><<<dim3(NJ / 64, 8), 128, 0, stream>>>(hbuf0, wr1h, GBR[1], xr, NJ, 128);
    gat_tail(GATT[1], GBIAS[1], 1);
    // layer 2: in = hbuf1 (K=128), mean over heads
    k_gemm_nt<128, 0><<<dim3(NJ / 64, 8), 128, 0, stream>>>(hbuf1, wl2h, GBL[2], xl, NJ, 128);
    k_gemm_nt<128, 0><<<dim3(NJ / 64, 8), 128, 0, stream>>>(hbuf1, wr2h, GBR[2], xr, NJ, 128);
    gat_tail(GATT[2], GBIAS[2], 2);

    // ---- head ----
    // h1 = relu(g @ hw1^T + hb1), M=8448 N=64 K=32
    k_gemm_nt<32, 1><<<dim3(NJ / 64, 4), 128, 0, stream>>>(gfh, hw1h, HB1, h1f, NJ, 64);
    k_cvt<<<(NJ * 64 + T - 1) / T, T, 0, stream>>>(h1f, h1h, NJ * 64);
    // o = h1 @ hw2^T + hb2, M=8448 N=4 K=64
    k_gemm_nt<64, 0><<<dim3(NJ / 64, 1), 128, 0, stream>>>(h1h, hw2h, HB2, o4, NJ, 4);
    // coords + sigmoid confidence -> d_out
    k_final<<<(NJ + T - 1) / T, T, 0, stream>>>(o4, (float*)d_out, NJ);
}
